// GraphAutoencoder_23751169147206
// MI455X (gfx1250) — matrix operand order repaired and checked by reading, not compiled, not run on hardware
//
#include <hip/hip_runtime.h>
#include <hip/hip_bf16.h>

typedef _Float16 half16 __attribute__((ext_vector_type(16)));
typedef float    float8 __attribute__((ext_vector_type(8)));

#define NNODES 50000
#define NHEADS 8

// ---------------- helpers ----------------

__device__ __forceinline__ float4 ldf4(const float* __restrict__ p) {
    return *reinterpret_cast<const float4*>(p);
}

__device__ __forceinline__ void cvt8(const float4& u0, const float4& u1,
                                     half16& v, int base) {
    v[base + 0] = (_Float16)u0.x; v[base + 1] = (_Float16)u0.y;
    v[base + 2] = (_Float16)u0.z; v[base + 3] = (_Float16)u0.w;
    v[base + 4] = (_Float16)u1.x; v[base + 5] = (_Float16)u1.y;
    v[base + 6] = (_Float16)u1.z; v[base + 7] = (_Float16)u1.w;
}

// order-preserving float <-> uint map for atomicMax-based segment max
__device__ __forceinline__ unsigned f2key(float f) {
    unsigned x = __float_as_uint(f);
    return (x & 0x80000000u) ? ~x : (x | 0x80000000u);
}
__device__ __forceinline__ float key2f(unsigned k) {
    return (k & 0x80000000u) ? __uint_as_float(k & 0x7FFFFFFFu) : __uint_as_float(~k);
}

__device__ __forceinline__ void get_edge(const int* __restrict__ ei, int E, int e,
                                         int& s, int& d) {
    if (e < E) { s = ei[e]; d = ei[E + e]; }
    else       { s = e - E; d = e - E; }      // self loops appended
}

// ---------------- kernels ----------------

// WT[n*K + k] = W[k*Nout + n]
__global__ void k_transpose(const float* __restrict__ W, float* __restrict__ WT,
                            int K, int Nout) {
    int t = blockIdx.x * blockDim.x + threadIdx.x;
    if (t >= K * Nout) return;
    int k = t / Nout, n = t % Nout;
    WT[n * K + k] = W[t];
}

// C[M,Nout] = A[M,K] @ B[K,Nout], B supplied transposed (BT[Nout,K]).
// One 16(M) x 64(N) supertile per wave32; A fragment reused across 4 B
// fragments -> 4 WMMAs per 32-wide K step, fully unrolled over K.
// All 20 float4 loads of a K-step are staged into distinct registers before
// conversion so they issue as one clause (launch_bounds gives VGPR headroom).
template<int K>
__global__ void __launch_bounds__(256, 1)
k_gemm_wmma(const float* __restrict__ A, const float* __restrict__ BT,
            float* __restrict__ C, int M, int Nout) {
    const int lane  = threadIdx.x;               // 0..31
    const int wave  = threadIdx.y;               // 0..7
    const int tileM = blockIdx.x;
    const int n0    = (blockIdx.y * 8 + wave) * 64;   // supertile col base
    if (n0 >= Nout) return;                      // wave-uniform

    const int half = lane >> 4;                  // 0 or 1
    const int l16  = lane & 15;
    const int row  = tileM * 16 + l16;

    const float* arow = A + (size_t)row * K;
    const float* b0r  = BT + (size_t)(n0 +  0 + l16) * K;
    const float* b1r  = BT + (size_t)(n0 + 16 + l16) * K;
    const float* b2r  = BT + (size_t)(n0 + 32 + l16) * K;
    const float* b3r  = BT + (size_t)(n0 + 48 + l16) * K;

    float8 acc0 = {0.f,0.f,0.f,0.f,0.f,0.f,0.f,0.f};
    float8 acc1 = acc0, acc2 = acc0, acc3 = acc0;

#pragma unroll
    for (int k0 = 0; k0 < K; k0 += 32) {
        // ---- stage all loads of this K-step (distinct regs -> one clause) ----
        // A 16x32 f16, contiguous k order (same as B): lanes0-15 hold K=0..15 ; lanes16-31 hold K=16..31
        float4 a0 = ldf4(arow + k0 + half * 16);
        float4 a1 = ldf4(arow + k0 + half * 16 + 4);
        float4 a2 = ldf4(arow + k0 + half * 16 + 8);
        float4 a3 = ldf4(arow + k0 + half * 16 + 12);
        // B 32x16 f16, contiguous k order (same as A): lanes0-15 hold K=0..15 ; lanes16-31 hold K=16..31
        float4 p0 = ldf4(b0r + k0 + half * 16);
        float4 p1 = ldf4(b0r + k0 + half * 16 + 4);
        float4 p2 = ldf4(b0r + k0 + half * 16 + 8);
        float4 p3 = ldf4(b0r + k0 + half * 16 + 12);
        float4 q0 = ldf4(b1r + k0 + half * 16);
        float4 q1 = ldf4(b1r + k0 + half * 16 + 4);
        float4 q2 = ldf4(b1r + k0 + half * 16 + 8);
        float4 q3 = ldf4(b1r + k0 + half * 16 + 12);
        float4 r0 = ldf4(b2r + k0 + half * 16);
        float4 r1 = ldf4(b2r + k0 + half * 16 + 4);
        float4 r2 = ldf4(b2r + k0 + half * 16 + 8);
        float4 r3 = ldf4(b2r + k0 + half * 16 + 12);
        float4 s0 = ldf4(b3r + k0 + half * 16);
        float4 s1 = ldf4(b3r + k0 + half * 16 + 4);
        float4 s2 = ldf4(b3r + k0 + half * 16 + 8);
        float4 s3 = ldf4(b3r + k0 + half * 16 + 12);

        // ---- convert ----
        half16 a, b0, b1, b2, b3;
        cvt8(a0, a1, a, 0);  cvt8(a2, a3, a, 8);
        cvt8(p0, p1, b0, 0); cvt8(p2, p3, b0, 8);
        cvt8(q0, q1, b1, 0); cvt8(q2, q3, b1, 8);
        cvt8(r0, r1, b2, 0); cvt8(r2, r3, b2, 8);
        cvt8(s0, s1, b3, 0); cvt8(s2, s3, b3, 8);

        // ---- 4 back-to-back WMMAs sharing the A fragment ----
        acc0 = __builtin_amdgcn_wmma_f32_16x16x32_f16(false, a, false, b0, (short)0, acc0, false, false);
        acc1 = __builtin_amdgcn_wmma_f32_16x16x32_f16(false, a, false, b1, (short)0, acc1, false, false);
        acc2 = __builtin_amdgcn_wmma_f32_16x16x32_f16(false, a, false, b2, (short)0, acc2, false, false);
        acc3 = __builtin_amdgcn_wmma_f32_16x16x32_f16(false, a, false, b3, (short)0, acc3, false, false);
    }

    // C/D layout: VGPR r -> row = r + 8*half, col = l16 (per 16-col tile)
#pragma unroll
    for (int r = 0; r < 8; ++r) {
        float* crow = C + (size_t)(tileM * 16 + half * 8 + r) * Nout + n0 + l16;
        crow[ 0] = acc0[r];
        crow[16] = acc1[r];
        crow[32] = acc2[r];
        crow[48] = acc3[r];
    }
}

// alpha_src[n,h] = sum_c h[n,h,c]*att_src[h,c] ; same for dst
__global__ void k_alphas(const float* __restrict__ h, const float* __restrict__ att_s,
                         const float* __restrict__ att_d, float* __restrict__ as,
                         float* __restrict__ ad, int N, int C) {
    int t = blockIdx.x * blockDim.x + threadIdx.x;
    if (t >= N * NHEADS) return;
    int n = t >> 3, hd = t & 7;
    const float* hp = h + (size_t)n * NHEADS * C + hd * C;
    const float* sp = att_s + hd * C;
    const float* dp = att_d + hd * C;
    float s = 0.f, d = 0.f;
    for (int c = 0; c < C; ++c) { float v = hp[c]; s += v * sp[c]; d += v * dp[c]; }
    as[t] = s; ad[t] = d;
}

__global__ void k_edge_max(const int* __restrict__ ei, int E, int Etot,
                           const float* __restrict__ as, const float* __restrict__ ad,
                           unsigned* __restrict__ mkey) {
    int t = blockIdx.x * blockDim.x + threadIdx.x;
    if (t >= Etot * NHEADS) return;
    int e = t >> 3, hd = t & 7, s, d;
    get_edge(ei, E, e, s, d);
    float x = as[s * NHEADS + hd] + ad[d * NHEADS + hd];
    x = (x >= 0.f) ? x : 0.2f * x;        // leaky_relu(0.2)
    atomicMax(&mkey[d * NHEADS + hd], f2key(x));
}

__global__ void k_edge_sum(const int* __restrict__ ei, int E, int Etot,
                           const float* __restrict__ as, const float* __restrict__ ad,
                           const unsigned* __restrict__ mkey, float* __restrict__ pbuf,
                           float* __restrict__ denom) {
    int t = blockIdx.x * blockDim.x + threadIdx.x;
    if (t >= Etot * NHEADS) return;
    int e = t >> 3, hd = t & 7, s, d;
    get_edge(ei, E, e, s, d);
    float x = as[s * NHEADS + hd] + ad[d * NHEADS + hd];
    x = (x >= 0.f) ? x : 0.2f * x;
    float p = __expf(x - key2f(mkey[d * NHEADS + hd]));
    pbuf[t] = p;
    atomicAdd(&denom[d * NHEADS + hd], p);
}

// acc[dst,h,c] += h[src,h,c] * p/(denom+eps) ; one thread per (edge, head, 4 channels)
__global__ void k_edge_aggr(const int* __restrict__ ei, int E, int Etot,
                            const float* __restrict__ h, const float* __restrict__ pbuf,
                            const float* __restrict__ denom, float* __restrict__ acc,
                            int C) {
    int cg = C >> 2;
    long long t = (long long)blockIdx.x * blockDim.x + threadIdx.x;
    long long total = (long long)Etot * NHEADS * cg;
    if (t >= total) return;
    int c  = (int)(t % cg) * 4;
    long long eh = t / cg;
    int hd = (int)(eh & 7);
    int e  = (int)(eh >> 3);
    int s, d;
    get_edge(ei, E, e, s, d);
    float al = pbuf[(size_t)e * NHEADS + hd] /
               (denom[d * NHEADS + hd] + 1e-16f);
    const float* hp = h + ((size_t)s * NHEADS + hd) * C + c;
    float4 hv = *reinterpret_cast<const float4*>(hp);
    float* ap = acc + ((size_t)d * NHEADS + hd) * C + c;
    atomicAdd(ap + 0, hv.x * al);
    atomicAdd(ap + 1, hv.y * al);
    atomicAdd(ap + 2, hv.z * al);
    atomicAdd(ap + 3, hv.w * al);
}

// out[n,c] = mean_h acc[n,h,c] + b[c] (+relu)
__global__ void k_finalize(const float* __restrict__ acc, const float* __restrict__ b,
                           float* __restrict__ out, int N, int C, int do_relu) {
    int t = blockIdx.x * blockDim.x + threadIdx.x;
    if (t >= N * C) return;
    int n = t / C, c = t % C;
    float s = 0.f;
#pragma unroll
    for (int hd = 0; hd < NHEADS; ++hd)
        s += acc[((size_t)n * NHEADS + hd) * C + c];
    s = s * (1.0f / NHEADS) + b[c];
    if (do_relu) s = fmaxf(s, 0.f);
    out[t] = s;
}

// ---------------- host ----------------

static inline int cdiv(long long a, int b) { return (int)((a + b - 1) / b); }

extern "C" void kernel_launch(void* const* d_in, const int* in_sizes, int n_in,
                              void* d_out, int out_size, void* d_ws, size_t ws_size,
                              hipStream_t stream) {
    const float* x    = (const float*)d_in[0];
    const int*   ei   = (const int*)d_in[1];
    // d_in[2] edge_attr: ignored (edge_dim=None)
    const float* W1   = (const float*)d_in[3];
    const float* as1  = (const float*)d_in[4];
    const float* ad1  = (const float*)d_in[5];
    const float* b1   = (const float*)d_in[6];
    const float* W2   = (const float*)d_in[7];
    const float* as2  = (const float*)d_in[8];
    const float* ad2  = (const float*)d_in[9];
    const float* b2   = (const float*)d_in[10];

    const int N    = NNODES;
    const int E    = in_sizes[1] / 2;
    const int Etot = E + N;

    // workspace carve-up (floats)
    float* ws = (float*)d_ws;
    size_t off = 0;
    float*    h     = ws + off; off += (size_t)N * 512;        // h1 / h2
    float*    acc   = ws + off; off += (size_t)N * 512;        // acc1 / acc2
    float*    x2    = ws + off; off += (size_t)N * 64;
    float*    as    = ws + off; off += (size_t)N * NHEADS;
    float*    ad    = ws + off; off += (size_t)N * NHEADS;
    unsigned* mkey  = (unsigned*)(ws + off); off += (size_t)N * NHEADS;
    float*    denom = ws + off; off += (size_t)N * NHEADS;
    float*    pbuf  = ws + off; off += (size_t)Etot * NHEADS;
    float*    WT    = ws + off; off += (size_t)128 * 512;

    const int BT = 256;
    dim3 gblock(32, 8);

    // ===================== layer 1 (F_IN=128 -> H x 64) =====================
    {
        const int K = 128, Nout = 512, C = 64;
        k_transpose<<<cdiv((long long)K * Nout, BT), BT, 0, stream>>>(W1, WT, K, Nout);
        dim3 ggrid(N / 16, cdiv(Nout, 64 * 8));   // (3125, 1)
        k_gemm_wmma<128><<<ggrid, gblock, 0, stream>>>(x, WT, h, N, Nout);
        k_alphas<<<cdiv((long long)N * NHEADS, BT), BT, 0, stream>>>(h, as1, ad1, as, ad, N, C);

        hipMemsetAsync(mkey,  0, (size_t)N * NHEADS * 4, stream);
        hipMemsetAsync(denom, 0, (size_t)N * NHEADS * 4, stream);
        hipMemsetAsync(acc,   0, (size_t)N * NHEADS * C * 4, stream);

        int eth = Etot * NHEADS;
        k_edge_max<<<cdiv(eth, BT), BT, 0, stream>>>(ei, E, Etot, as, ad, mkey);
        k_edge_sum<<<cdiv(eth, BT), BT, 0, stream>>>(ei, E, Etot, as, ad, mkey, pbuf, denom);
        long long ath = (long long)Etot * NHEADS * (C / 4);
        k_edge_aggr<<<cdiv(ath, BT), BT, 0, stream>>>(ei, E, Etot, h, pbuf, denom, acc, C);
        k_finalize<<<cdiv((long long)N * C, BT), BT, 0, stream>>>(acc, b1, x2, N, C, 1);
    }

    // ===================== layer 2 (64 -> H x 32) =====================
    {
        const int K = 64, Nout = 256, C = 32;
        k_transpose<<<cdiv((long long)K * Nout, BT), BT, 0, stream>>>(W2, WT, K, Nout);
        dim3 ggrid(N / 16, cdiv(Nout, 64 * 8));   // (3125, 1); waves 4..7 idle-exit
        k_gemm_wmma<64><<<ggrid, gblock, 0, stream>>>(x2, WT, h, N, Nout);
        k_alphas<<<cdiv((long long)N * NHEADS, BT), BT, 0, stream>>>(h, as2, ad2, as, ad, N, C);

        hipMemsetAsync(mkey,  0, (size_t)N * NHEADS * 4, stream);
        hipMemsetAsync(denom, 0, (size_t)N * NHEADS * 4, stream);
        hipMemsetAsync(acc,   0, (size_t)N * NHEADS * C * 4, stream);

        int eth = Etot * NHEADS;
        k_edge_max<<<cdiv(eth, BT), BT, 0, stream>>>(ei, E, Etot, as, ad, mkey);
        k_edge_sum<<<cdiv(eth, BT), BT, 0, stream>>>(ei, E, Etot, as, ad, mkey, pbuf, denom);
        long long ath = (long long)Etot * NHEADS * (C / 4);
        k_edge_aggr<<<cdiv(ath, BT), BT, 0, stream>>>(ei, E, Etot, h, pbuf, denom, acc, C);
        k_finalize<<<cdiv((long long)N * C, BT), BT, 0, stream>>>(acc, b2, (float*)d_out, N, C, 0);
    }
    (void)n_in; (void)out_size; (void)ws_size; (void)in_sizes;
}
